// GIN_94489280747
// MI455X (gfx1250) — matrix operand order repaired and checked by reading, not compiled, not run on hardware
//
#include <hip/hip_runtime.h>

#define NN 50000
#define NE 600000
#define D 128
#define NODE_F (NN * D)
#define BR 128          // rows per block in GEMM kernels
#define GTHREADS 256    // 8 waves per block

typedef __attribute__((ext_vector_type(16))) __bf16 v16bf;
typedef __attribute__((ext_vector_type(8)))  float  v8f;

union Frag16 { v16bf v; uint4 u[2]; };
union Pack4  { __bf16 b[4]; uint2 u; };

// ---------------- zero agg buffer ----------------
__global__ void gin_zero_kernel(float* __restrict__ p, int n4) {
    int i = blockIdx.x * blockDim.x + threadIdx.x;
    if (i < n4) ((float4*)p)[i] = make_float4(0.f, 0.f, 0.f, 0.f);
}

// ---------------- one-time weight prep: fp32 -> bf16, transposed [n][k] ----------------
// wbuf layout: [W1_l0, W2_l0, W1_l1, W2_l1, W1_l2, W2_l2, Wf_c0..Wf_c3], each D*D bf16
__global__ void gin_prep_weights(const float* __restrict__ W1,
                                 const float* __restrict__ W2,
                                 const float* __restrict__ Wf,
                                 __bf16* __restrict__ wbuf) {
    int idx = blockIdx.x * blockDim.x + threadIdx.x;
    if (idx >= 10 * D * D) return;
    int m   = idx >> 14;              // which D*D matrix (16384 = 2^14)
    int rem = idx & (D * D - 1);
    int k   = rem >> 7, n = rem & 127;
    float v;
    if (m < 6) {
        int l = m >> 1;
        const float* W = (m & 1) ? W2 : W1;
        v = W[l * D * D + k * D + n];
    } else {
        int c = m - 6;
        v = Wf[(size_t)(c * D + k) * D + n];
    }
    wbuf[(m << 14) + n * D + k] = (__bf16)v;
}

// ---------------- edge scatter-sum: agg[dst] += h[src] ----------------
__global__ void gin_agg_kernel(const float* __restrict__ h,
                               const long long* __restrict__ eidx,
                               float* __restrict__ agg) {
    int gid = blockIdx.x * blockDim.x + threadIdx.x;
    int e   = __builtin_amdgcn_readfirstlane(gid >> 5);  // wave-uniform edge id -> SGPR
    if (e >= NE) return;
    int lane = threadIdx.x & 31;       // each lane: 4 consecutive features
    int src  = (int)eidx[e];           // uniform address -> s_load
    int dst  = (int)eidx[NE + e];
    const float4 v = *(const float4*)(h + (size_t)src * D + lane * 4);
    float* o = agg + (size_t)dst * D + lane * 4;
    unsafeAtomicAdd(o + 0, v.x);
    unsafeAtomicAdd(o + 1, v.y);
    unsafeAtomicAdd(o + 2, v.z);
    unsafeAtomicAdd(o + 3, v.w);
}

// A fragment (16x32 bf16, MxK) from row-major LDS tile [rows][D]
__device__ __forceinline__ Frag16 load_afrag(const __bf16* base, int row, int kf, int hi) {
    Frag16 a;
    const __bf16* p = base + row * D + kf * 32 + hi * 8;  // K = kf*32 + {0..7 | 8..15}
    a.u[0] = *(const uint4*)p;          // K + 0..7
    a.u[1] = *(const uint4*)(p + 16);   // K + 16..23
    return a;
}

// B fragment (32x16 bf16, KxN) from LDS weights stored TRANSPOSED: sW[n*D + k]
__device__ __forceinline__ Frag16 load_bfrag(const __bf16* sWt, int col, int kf, int hi) {
    Frag16 b;
    const __bf16* p = sWt + col * D + kf * 32 + hi * 8;  // K = kf*32 + {0..7 | 8..15}
    b.u[0] = *(const uint4*)p;          // K + 0..7
    b.u[1] = *(const uint4*)(p + 16);   // K + 16..23
    return b;
}

// copy D*D bf16 matrix global -> LDS as raw b128 moves
__device__ __forceinline__ void stage_weights(const __bf16* __restrict__ g,
                                              __bf16* __restrict__ s, int tid) {
    const uint4* src = (const uint4*)g;
    uint4*       dst = (uint4*)s;
    for (int i = tid; i < (D * D) / 8; i += GTHREADS) dst[i] = src[i];
}

// ---------------- one GIN layer: hout = relu(((1+eps)h+agg) W1 + b1) W2 + b2 ----------------
__launch_bounds__(GTHREADS, 1)
__global__ void gin_mlp_kernel(const float* __restrict__ h,
                               const float* __restrict__ agg,
                               const float* __restrict__ eps_arr, int layer,
                               const __bf16* __restrict__ W1t,
                               const float* __restrict__ b1,
                               const __bf16* __restrict__ W2t,
                               const float* __restrict__ b2,
                               float* __restrict__ hout) {
    __shared__ __align__(16) __bf16 sW1[D * D];
    __shared__ __align__(16) __bf16 sW2[D * D];
    __shared__ __align__(16) __bf16 sZ[BR * D];
    __shared__ __align__(16) __bf16 sH[BR * D];

    const int tid = threadIdx.x;
    const int r0  = blockIdx.x * BR;
    const float eps1 = 1.0f + eps_arr[layer];

    stage_weights(W1t, sW1, tid);
    stage_weights(W2t, sW2, tid);

    // stage Z = (1+eps)*h + agg  (zero-pad past NN), float4 -> packed bf16x4
    for (int i = tid; i < (BR * D) / 4; i += GTHREADS) {
        int row = i >> 5;                 // D/4 = 32 float4 per row
        int c4  = (i & 31) * 4;
        int r   = r0 + row;
        Pack4 pk;
        if (r < NN) {
            size_t off = (size_t)r * D + c4;
            float4 hv = *(const float4*)(h + off);
            float4 av = *(const float4*)(agg + off);
            pk.b[0] = (__bf16)(eps1 * hv.x + av.x);
            pk.b[1] = (__bf16)(eps1 * hv.y + av.y);
            pk.b[2] = (__bf16)(eps1 * hv.z + av.z);
            pk.b[3] = (__bf16)(eps1 * hv.w + av.w);
        } else {
            pk.b[0] = pk.b[1] = pk.b[2] = pk.b[3] = (__bf16)0.f;
        }
        *(uint2*)&sZ[row * D + c4] = pk.u;
    }
    __syncthreads();

    const int lane = tid & 31;
    const int wv   = tid >> 5;
    const int rw   = wv * 16;                 // wave's 16-row tile in block
    const int hi   = (lane >= 16) ? 1 : 0;
    const int an   = lane & 15;

    // ---- GEMM1: sH = relu(sZ @ W1 + b1) ----
    Frag16 A[4];
    for (int kf = 0; kf < 4; ++kf) A[kf] = load_afrag(sZ, rw + an, kf, hi);
    for (int nt = 0; nt < 8; ++nt) {
        int col = nt * 16 + an;
        float bias = b1[col];
        v8f acc = { bias, bias, bias, bias, bias, bias, bias, bias };
        for (int kf = 0; kf < 4; ++kf) {
            Frag16 B = load_bfrag(sW1, col, kf, hi);
            acc = __builtin_amdgcn_wmma_f32_16x16x32_bf16(
                      false, A[kf].v, false, B.v, (short)0, acc, false, false);
        }
        for (int i = 0; i < 8; ++i) {
            float v = acc[i];
            v = v > 0.f ? v : 0.f;
            sH[(rw + i + hi * 8) * D + col] = (__bf16)v;
        }
    }
    __syncthreads();

    // ---- GEMM2: hout = sH @ W2 + b2 ----
    Frag16 A2[4];
    for (int kf = 0; kf < 4; ++kf) A2[kf] = load_afrag(sH, rw + an, kf, hi);
    for (int nt = 0; nt < 8; ++nt) {
        int col = nt * 16 + an;
        float bias = b2[col];
        v8f acc = { bias, bias, bias, bias, bias, bias, bias, bias };
        for (int kf = 0; kf < 4; ++kf) {
            Frag16 B = load_bfrag(sW2, col, kf, hi);
            acc = __builtin_amdgcn_wmma_f32_16x16x32_bf16(
                      false, A2[kf].v, false, B.v, (short)0, acc, false, false);
        }
        for (int i = 0; i < 8; ++i) {
            int r = r0 + rw + i + hi * 8;
            if (r < NN) hout[(size_t)r * D + col] = acc[i];
        }
    }
}

// ---------------- final: out = cat(x,h1,h2,h3) @ Wf + bf ----------------
__launch_bounds__(GTHREADS, 1)
__global__ void gin_final_kernel(const float* __restrict__ x,
                                 const float* __restrict__ h1,
                                 const float* __restrict__ h2,
                                 const float* __restrict__ h3,
                                 const __bf16* __restrict__ Wft,  // 4 chunks, bf16 transposed
                                 const float* __restrict__ bfv,
                                 float* __restrict__ out) {
    __shared__ __align__(16) __bf16 sW[D * D];
    __shared__ __align__(16) __bf16 sA[BR * D];

    const int tid  = threadIdx.x;
    const int r0   = blockIdx.x * BR;
    const int lane = tid & 31;
    const int wv   = tid >> 5;
    const int rw   = wv * 16;
    const int hi   = (lane >= 16) ? 1 : 0;
    const int an   = lane & 15;

    v8f acc[8];
    for (int nt = 0; nt < 8; ++nt) {
        float bias = bfv[nt * 16 + an];
        for (int i = 0; i < 8; ++i) acc[nt][i] = bias;
    }

    const float* srcs[4] = { x, h1, h2, h3 };
    for (int c = 0; c < 4; ++c) {
        __syncthreads();   // protect LDS reuse across chunk iterations
        const float* S = srcs[c];
        stage_weights(Wft + c * D * D, sW, tid);
        for (int i = tid; i < (BR * D) / 4; i += GTHREADS) {
            int row = i >> 5;
            int c4  = (i & 31) * 4;
            int r   = r0 + row;
            Pack4 pk;
            if (r < NN) {
                float4 v = *(const float4*)(S + (size_t)r * D + c4);
                pk.b[0] = (__bf16)v.x; pk.b[1] = (__bf16)v.y;
                pk.b[2] = (__bf16)v.z; pk.b[3] = (__bf16)v.w;
                if (c < 3)  // warm L2/L0 for next chunk's tile (global_prefetch_b8)
                    __builtin_prefetch(srcs[c + 1] + (size_t)r * D + c4, 0, 3);
            } else {
                pk.b[0] = pk.b[1] = pk.b[2] = pk.b[3] = (__bf16)0.f;
            }
            *(uint2*)&sA[row * D + c4] = pk.u;
        }
        __syncthreads();

        Frag16 A[4];
        for (int kf = 0; kf < 4; ++kf) A[kf] = load_afrag(sA, rw + an, kf, hi);
        for (int nt = 0; nt < 8; ++nt) {
            int col = nt * 16 + an;
            for (int kf = 0; kf < 4; ++kf) {
                Frag16 B = load_bfrag(sW, col, kf, hi);
                acc[nt] = __builtin_amdgcn_wmma_f32_16x16x32_bf16(
                              false, A[kf].v, false, B.v, (short)0, acc[nt], false, false);
            }
        }
    }

    for (int nt = 0; nt < 8; ++nt) {
        int col = nt * 16 + an;
        for (int i = 0; i < 8; ++i) {
            int r = r0 + rw + i + hi * 8;
            if (r < NN) out[(size_t)r * D + col] = acc[nt][i];
        }
    }
}

extern "C" void kernel_launch(void* const* d_in, const int* in_sizes, int n_in,
                              void* d_out, int out_size, void* d_ws, size_t ws_size,
                              hipStream_t stream) {
    const float*     x    = (const float*)d_in[0];
    const long long* eidx = (const long long*)d_in[1];   // int64 per reference
    const float*     eps  = (const float*)d_in[2];
    const float*     W1   = (const float*)d_in[3];
    const float*     b1   = (const float*)d_in[4];
    const float*     W2   = (const float*)d_in[5];
    const float*     b2   = (const float*)d_in[6];
    const float*     Wf   = (const float*)d_in[7];
    const float*     bfv  = (const float*)d_in[8];
    float*           out  = (float*)d_out;

    float*  agg  = (float*)d_ws;             // NODE_F floats
    float*  h1   = agg + NODE_F;
    float*  h2   = h1 + NODE_F;
    float*  h3   = h2 + NODE_F;
    __bf16* wbuf = (__bf16*)(h3 + NODE_F);   // 10 * D*D bf16 (320 KB)

    const float* hin[3]  = { x, h1, h2 };
    float*       hout[3] = { h1, h2, h3 };

    const int zgrid = (NODE_F / 4 + 255) / 256;
    const int agrid = (int)(((long long)NE * 32 + 255) / 256);
    const int ggrid = (NN + BR - 1) / BR;
    const int pgrid = (10 * D * D + 255) / 256;

    gin_prep_weights<<<pgrid, 256, 0, stream>>>(W1, W2, Wf, wbuf);

    for (int l = 0; l < 3; ++l) {
        gin_zero_kernel<<<zgrid, 256, 0, stream>>>(agg, NODE_F / 4);
        gin_agg_kernel<<<agrid, 256, 0, stream>>>(hin[l], eidx, agg);
        gin_mlp_kernel<<<ggrid, GTHREADS, 0, stream>>>(
            hin[l], agg, eps, l,
            wbuf + (2 * l + 0) * D * D, b1 + l * D,
            wbuf + (2 * l + 1) * D * D, b2 + l * D,
            hout[l]);
    }
    gin_final_kernel<<<ggrid, GTHREADS, 0, stream>>>(x, h1, h2, h3,
                                                     wbuf + 6 * D * D, bfv, out);
}
